// MultiHeadAttention_35107062677624
// MI455X (gfx1250) — matrix operand order repaired and checked by reading, not compiled, not run on hardware
//
#include <hip/hip_runtime.h>
#include <math.h>

typedef __attribute__((ext_vector_type(16))) _Float16 v16h;
typedef __attribute__((ext_vector_type(8)))  _Float16 v8h;
typedef __attribute__((ext_vector_type(8)))  float    v8f;
typedef __attribute__((ext_vector_type(4)))  float    v4f;

#define DIM   512
#define SEQ   1024
#define BATCH 8
#define HEADS 8
#define DHEAD 64
#define ROWS_TOTAL (BATCH*SEQ)   // 8192

// Build a v16h WMMA fragment from two contiguous 16-byte LDS chunks.
static __device__ __forceinline__ v16h make_frag(const _Float16* p0, const _Float16* p1) {
    v8h lo = *(const v8h*)p0;
    v8h hi = *(const v8h*)p1;
    return __builtin_shufflevector(lo, hi, 0,1,2,3,4,5,6,7,8,9,10,11,12,13,14,15);
}

// CDNA5 async global->LDS copy: 16 bytes per lane, tracked by ASYNCcnt.
// saddr form: global_load_async_to_lds_b128 vdst(lds byte addr), vaddr(u32 off), saddr
static __device__ __forceinline__ void async_copy_b128(unsigned long long gbase,
                                                       unsigned int voff_bytes,
                                                       _Float16* lds_dst) {
    unsigned int l = (unsigned int)(uintptr_t)lds_dst;   // LDS byte offset (low 32 bits of generic ptr)
    asm volatile("global_load_async_to_lds_b128 %0, %1, %2"
                 :: "v"(l), "v"(voff_bytes), "s"(gbase) : "memory");
}
static __device__ __forceinline__ void wait_async_all() {
#if __has_builtin(__builtin_amdgcn_s_wait_asynccnt)
    __builtin_amdgcn_s_wait_asynccnt(0);
#else
    asm volatile("s_wait_asynccnt 0" ::: "memory");
#endif
}

// ---------------------------------------------------------------------------
// Weight f32 -> f16 pack
// ---------------------------------------------------------------------------
__global__ void pack_f16_kernel(const float* __restrict__ src, _Float16* __restrict__ dst, int n) {
    int i = blockIdx.x * blockDim.x + threadIdx.x;
    int stride = gridDim.x * blockDim.x;
    for (; i < n; i += stride) dst[i] = (_Float16)src[i];
}

// ---------------------------------------------------------------------------
// LayerNorm over channels with implicit [b][c][s] -> [b][s][c] transpose.
// One wave per 512-wide row; 16 channels per lane; wave32 shfl reductions.
// ---------------------------------------------------------------------------
__global__ __launch_bounds__(256) void ln_q_kernel(const float* __restrict__ x,
                                                   const float* __restrict__ g,
                                                   const float* __restrict__ bb,
                                                   _Float16* __restrict__ out) {
    const int wv = threadIdx.x >> 5, lane = threadIdx.x & 31;
    const int R = blockIdx.x * 8 + wv;
    const int b = R >> 10, s = R & 1023;
    const float* xp = x + (size_t)b * DIM * SEQ + s;
    float vals[16];
    float sum = 0.f, sq = 0.f;
#pragma unroll
    for (int j = 0; j < 16; ++j) {
        int c = j * 32 + lane;
        float v = xp[(size_t)c * SEQ];
        vals[j] = v; sum += v; sq += v * v;
    }
#pragma unroll
    for (int m = 1; m < 32; m <<= 1) { sum += __shfl_xor(sum, m, 32); sq += __shfl_xor(sq, m, 32); }
    float mean = sum * (1.f / 512.f);
    float var  = sq  * (1.f / 512.f) - mean * mean;
    float rstd = rsqrtf(var + 1e-5f);
    _Float16* op = out + (size_t)R * DIM;
#pragma unroll
    for (int j = 0; j < 16; ++j) {
        int c = j * 32 + lane;
        op[c] = (_Float16)((vals[j] - mean) * rstd * g[c] + bb[c]);
    }
}

// kn = LN(k); vn = LN(kn) (reference quirk) in one pass.
__global__ __launch_bounds__(256) void ln_kv_kernel(const float* __restrict__ x,
                                                    const float* __restrict__ g,
                                                    const float* __restrict__ bb,
                                                    _Float16* __restrict__ outK,
                                                    _Float16* __restrict__ outV) {
    const int wv = threadIdx.x >> 5, lane = threadIdx.x & 31;
    const int R = blockIdx.x * 8 + wv;
    const int b = R >> 10, s = R & 1023;
    const float* xp = x + (size_t)b * DIM * SEQ + s;
    float vals[16];
    float sum = 0.f, sq = 0.f;
#pragma unroll
    for (int j = 0; j < 16; ++j) {
        int c = j * 32 + lane;
        float v = xp[(size_t)c * SEQ];
        vals[j] = v; sum += v; sq += v * v;
    }
#pragma unroll
    for (int m = 1; m < 32; m <<= 1) { sum += __shfl_xor(sum, m, 32); sq += __shfl_xor(sq, m, 32); }
    float mean = sum * (1.f / 512.f);
    float var  = sq  * (1.f / 512.f) - mean * mean;
    float rstd = rsqrtf(var + 1e-5f);
    _Float16* opK = outK + (size_t)R * DIM;
    _Float16* opV = outV + (size_t)R * DIM;
    float kn[16];
    float sum2 = 0.f, sq2 = 0.f;
#pragma unroll
    for (int j = 0; j < 16; ++j) {
        int c = j * 32 + lane;
        float y = (vals[j] - mean) * rstd * g[c] + bb[c];
        kn[j] = y; sum2 += y; sq2 += y * y;
        opK[c] = (_Float16)y;
    }
#pragma unroll
    for (int m = 1; m < 32; m <<= 1) { sum2 += __shfl_xor(sum2, m, 32); sq2 += __shfl_xor(sq2, m, 32); }
    float mean2 = sum2 * (1.f / 512.f);
    float var2  = sq2  * (1.f / 512.f) - mean2 * mean2;
    float rstd2 = rsqrtf(var2 + 1e-5f);
#pragma unroll
    for (int j = 0; j < 16; ++j) {
        int c = j * 32 + lane;
        opV[c] = (_Float16)((kn[j] - mean2) * rstd2 * g[c] + bb[c]);
    }
}

// ---------------------------------------------------------------------------
// QKV projection GEMM: C[8192x512](f16) = A[8192x512] * B[512x512] (both f16)
// 8 waves (4x2), tile 128x64, BK=32. Async A-tile copies; LDS-staged epilogue.
// ---------------------------------------------------------------------------
__global__ __launch_bounds__(256) void gemm_qkv_kernel(const _Float16* __restrict__ A,
                                                       const _Float16* __restrict__ B,
                                                       _Float16* __restrict__ C) {
    __shared__ __align__(16) _Float16 smem[128 * 72];   // As/Bs alias + epilogue staging
    _Float16* As = smem;                                // [128][40]
    _Float16* Bs = smem + 128 * 40;                     // [64][40]  (B transposed: [n][k])
    const int t = threadIdx.x;
    const int wv = t >> 5, lane = t & 31;
    const int halfsel = lane >> 4, ln16 = lane & 15;
    const int wm = wv >> 1, wn = wv & 1;
    const int m0 = blockIdx.y * 128, n0 = blockIdx.x * 64;
    const unsigned long long abase = (unsigned long long)(uintptr_t)A;

    v8f acc[2][2];
    for (int mi = 0; mi < 2; ++mi)
        for (int ni = 0; ni < 2; ++ni)
            for (int e = 0; e < 8; ++e) acc[mi][ni][e] = 0.f;

    for (int k0 = 0; k0 < DIM; k0 += 32) {
        __syncthreads();
        // A tile 128x32: async global->LDS, 16B per lane, 2 issues per thread
#pragma unroll
        for (int it = 0; it < 2; ++it) {
            int gidx = t + it * 256;
            int row = gidx >> 2, col8 = (gidx & 3) * 8;
            async_copy_b128(abase, (unsigned)(((m0 + row) * DIM + k0 + col8) * 2),
                            &As[row * 40 + col8]);
        }
        // B tile 32x64 -> transposed LDS [n][k]
        {
            int kk = t >> 3, nn8 = (t & 7) * 8;
            v8h bb8 = *(const v8h*)&B[(size_t)(k0 + kk) * DIM + n0 + nn8];
#pragma unroll
            for (int i = 0; i < 8; ++i) Bs[(nn8 + i) * 40 + kk] = bb8[i];
            if (k0 + 32 < DIM)
                __builtin_prefetch(&B[(size_t)(k0 + 32 + kk) * DIM + n0 + nn8], 0, 1);
        }
        wait_async_all();
        __syncthreads();

        v16h afrag[2], bfrag[2];
#pragma unroll
        for (int mi = 0; mi < 2; ++mi) {
            int ar = wm * 32 + mi * 16 + ln16;
            afrag[mi] = make_frag(&As[ar * 40 + 8 * halfsel],
                                  &As[ar * 40 + 16 + 8 * halfsel]);
        }
#pragma unroll
        for (int ni = 0; ni < 2; ++ni) {
            int bc = wn * 32 + ni * 16 + ln16;
            bfrag[ni] = make_frag(&Bs[bc * 40 + 8 * halfsel],
                                  &Bs[bc * 40 + 16 + 8 * halfsel]);
        }
#pragma unroll
        for (int mi = 0; mi < 2; ++mi)
#pragma unroll
            for (int ni = 0; ni < 2; ++ni)
                acc[mi][ni] = __builtin_amdgcn_wmma_f32_16x16x32_f16(
                    false, afrag[mi], false, bfrag[ni], (short)0, acc[mi][ni], false, false);
    }

    // Epilogue: stage f16 tile [128][72] in LDS, then coalesced b128 stores.
    __syncthreads();
#pragma unroll
    for (int mi = 0; mi < 2; ++mi)
#pragma unroll
        for (int ni = 0; ni < 2; ++ni)
#pragma unroll
            for (int r = 0; r < 8; ++r)
                smem[(wm * 32 + mi * 16 + r + 8 * halfsel) * 72 + wn * 32 + ni * 16 + ln16] =
                    (_Float16)acc[mi][ni][r];
    __syncthreads();
#pragma unroll
    for (int v = 0; v < 4; ++v) {
        int g = t + v * 256;
        int row = g >> 3, col8 = (g & 7) * 8;
        *(v8h*)&C[(size_t)(m0 + row) * DIM + n0 + col8] = *(const v8h*)&smem[row * 72 + col8];
    }
}

// ---------------------------------------------------------------------------
// Output projection: Out[b][c][s] = (A[8192x512] * Wo[512x512])[b*1024+s][c] + bo[c]
// f32 result staged transposed in LDS -> contiguous-in-s b128 stores.
// ---------------------------------------------------------------------------
__global__ __launch_bounds__(256) void gemm_out_kernel(const _Float16* __restrict__ A,
                                                       const _Float16* __restrict__ B,
                                                       const float* __restrict__ bias,
                                                       float* __restrict__ Out) {
    __shared__ __align__(16) float smemf[64 * 132];     // 33792 B; aliases As/Bs
    _Float16* As = (_Float16*)smemf;                    // [128][40]
    _Float16* Bs = (_Float16*)smemf + 128 * 40;         // [64][40]
    const int t = threadIdx.x;
    const int wv = t >> 5, lane = t & 31;
    const int halfsel = lane >> 4, ln16 = lane & 15;
    const int wm = wv >> 1, wn = wv & 1;
    const int m0 = blockIdx.y * 128, n0 = blockIdx.x * 64;
    const unsigned long long abase = (unsigned long long)(uintptr_t)A;

    v8f acc[2][2];
    for (int mi = 0; mi < 2; ++mi)
        for (int ni = 0; ni < 2; ++ni)
            for (int e = 0; e < 8; ++e) acc[mi][ni][e] = 0.f;

    for (int k0 = 0; k0 < DIM; k0 += 32) {
        __syncthreads();
#pragma unroll
        for (int it = 0; it < 2; ++it) {
            int gidx = t + it * 256;
            int row = gidx >> 2, col8 = (gidx & 3) * 8;
            async_copy_b128(abase, (unsigned)(((m0 + row) * DIM + k0 + col8) * 2),
                            &As[row * 40 + col8]);
        }
        {
            int kk = t >> 3, nn8 = (t & 7) * 8;
            v8h bb8 = *(const v8h*)&B[(size_t)(k0 + kk) * DIM + n0 + nn8];
#pragma unroll
            for (int i = 0; i < 8; ++i) Bs[(nn8 + i) * 40 + kk] = bb8[i];
            if (k0 + 32 < DIM)
                __builtin_prefetch(&B[(size_t)(k0 + 32 + kk) * DIM + n0 + nn8], 0, 1);
        }
        wait_async_all();
        __syncthreads();

        v16h afrag[2], bfrag[2];
#pragma unroll
        for (int mi = 0; mi < 2; ++mi) {
            int ar = wm * 32 + mi * 16 + ln16;
            afrag[mi] = make_frag(&As[ar * 40 + 8 * halfsel],
                                  &As[ar * 40 + 16 + 8 * halfsel]);
        }
#pragma unroll
        for (int ni = 0; ni < 2; ++ni) {
            int bc = wn * 32 + ni * 16 + ln16;
            bfrag[ni] = make_frag(&Bs[bc * 40 + 8 * halfsel],
                                  &Bs[bc * 40 + 16 + 8 * halfsel]);
        }
#pragma unroll
        for (int mi = 0; mi < 2; ++mi)
#pragma unroll
            for (int ni = 0; ni < 2; ++ni)
                acc[mi][ni] = __builtin_amdgcn_wmma_f32_16x16x32_f16(
                    false, afrag[mi], false, bfrag[ni], (short)0, acc[mi][ni], false, false);
    }

    // Epilogue: stage f32 transposed [col][row], then stores contiguous in s.
    __syncthreads();
#pragma unroll
    for (int mi = 0; mi < 2; ++mi)
#pragma unroll
        for (int ni = 0; ni < 2; ++ni)
#pragma unroll
            for (int r = 0; r < 8; ++r) {
                int lrow = wm * 32 + mi * 16 + r + 8 * halfsel;   // 0..127 (s)
                int lcol = wn * 32 + ni * 16 + ln16;              // 0..63  (c)
                smemf[lcol * 132 + lrow] = acc[mi][ni][r];
            }
    __syncthreads();
    const int ob = m0 >> 10, s0 = m0 & 1023;
#pragma unroll
    for (int v = 0; v < 8; ++v) {
        int g = t + v * 256;            // 0..2047
        int col = g >> 5, q4 = (g & 31) * 4;
        v4f val = *(const v4f*)&smemf[col * 132 + q4];
        float bv = bias[n0 + col];
        v4f r; r[0] = val[0] + bv; r[1] = val[1] + bv; r[2] = val[2] + bv; r[3] = val[3] + bv;
        *(v4f*)&Out[(size_t)ob * DIM * SEQ + (size_t)(n0 + col) * SEQ + s0 + q4] = r;
    }
}

// ---------------------------------------------------------------------------
// Flash attention: one block per (b,h, 128 q rows). 8 waves x 16 rows.
// Async Q/K tile copies; online softmax; WMMA f16 for QK^T and PV.
// ---------------------------------------------------------------------------
__global__ __launch_bounds__(256) void attn_kernel(
        const _Float16* __restrict__ Qp, const _Float16* __restrict__ Kp,
        const _Float16* __restrict__ Vp, _Float16* __restrict__ Op) {
    __shared__ __align__(16) _Float16 Qs[128 * 72];     // Q block; reused for epilogue staging
    __shared__ __align__(16) _Float16 Ks[32 * 72];
    __shared__ __align__(16) _Float16 Vt[64 * 40];      // V transposed: [d][j]
    __shared__ __align__(16) _Float16 Ps[8 * 16 * 40];  // per-wave P staging
    const int t = threadIdx.x;
    const int wv = t >> 5, lane = t & 31;
    const int halfsel = lane >> 4, ln16 = lane & 15;
    const int bh = blockIdx.y;
    const int b = bh >> 3, h = bh & 7;
    const int q0 = blockIdx.x * 128;
    const size_t rowbase = (size_t)b * SEQ;
    const size_t cbase = (size_t)h * DHEAD;
    const unsigned long long qbase = (unsigned long long)(uintptr_t)Qp;
    const unsigned long long kbase = (unsigned long long)(uintptr_t)Kp;

    // Resident Q block (128 x 64) via async copies
#pragma unroll
    for (int v = 0; v < 4; ++v) {
        int g = t + v * 256;
        int row = g >> 3, col8 = (g & 7) * 8;
        async_copy_b128(qbase,
            (unsigned)(((unsigned)(rowbase + q0 + row) * DIM + (unsigned)cbase + col8) * 2),
            &Qs[row * 72 + col8]);
    }
    wait_async_all();
    __syncthreads();
    v16h aq[2];
    {
        int qr = wv * 16 + ln16;
#pragma unroll
        for (int kk = 0; kk < 2; ++kk)
            aq[kk] = make_frag(&Qs[qr * 72 + kk * 32 + 8 * halfsel],
                               &Qs[qr * 72 + kk * 32 + 16 + 8 * halfsel]);
    }

    float mrow[8], lrow[8];
    v8f o_acc[4];
#pragma unroll
    for (int r = 0; r < 8; ++r) { mrow[r] = -INFINITY; lrow[r] = 0.f; }
    for (int dt = 0; dt < 4; ++dt)
        for (int e = 0; e < 8; ++e) o_acc[dt][e] = 0.f;
    const int pbase = wv * 16 * 40;

    for (int j0 = 0; j0 < SEQ; j0 += 32) {
        __syncthreads();  // previous-iter LDS reads done
        {
            int row = t >> 3, col8 = (t & 7) * 8;
            // K tile: async direct copy (B-operand layout == row-major K)
            async_copy_b128(kbase,
                (unsigned)(((unsigned)(rowbase + j0 + row) * DIM + (unsigned)cbase + col8) * 2),
                &Ks[row * 72 + col8]);
            // V tile: manual transposed store [d][j]
            v8h vvv = *(const v8h*)&Vp[(rowbase + j0 + row) * DIM + cbase + col8];
#pragma unroll
            for (int i = 0; i < 8; ++i) Vt[(col8 + i) * 40 + row] = vvv[i];
        }
        wait_async_all();
        __syncthreads();

        // S = Q * K^T : two 16x16 tiles per wave, K-dim 64 (2 WMMA each)
        v8f sacc[2];
#pragma unroll
        for (int jt = 0; jt < 2; ++jt) {
            for (int e = 0; e < 8; ++e) sacc[jt][e] = 0.f;
            int jl = jt * 16 + ln16;
#pragma unroll
            for (int kk = 0; kk < 2; ++kk) {
                v16h bk = make_frag(&Ks[jl * 72 + kk * 32 + 8 * halfsel],
                                    &Ks[jl * 72 + kk * 32 + 16 + 8 * halfsel]);
                sacc[jt] = __builtin_amdgcn_wmma_f32_16x16x32_f16(
                    false, aq[kk], false, bk, (short)0, sacc[jt], false, false);
            }
        }

        // Online softmax; row = vgpr r + 8*halfsel, reductions within 16-lane half.
        float alpha[8];
#pragma unroll
        for (int r = 0; r < 8; ++r) {
            float s0 = sacc[0][r] * 0.125f;
            float s1 = sacc[1][r] * 0.125f;
            float mx = fmaxf(s0, s1);
#pragma unroll
            for (int m = 1; m < 16; m <<= 1) mx = fmaxf(mx, __shfl_xor(mx, m, 32));
            float mnew = fmaxf(mrow[r], mx);
            alpha[r] = __expf(mrow[r] - mnew);
            mrow[r] = mnew;
            float p0 = __expf(s0 - mnew);
            float p1 = __expf(s1 - mnew);
            sacc[0][r] = p0; sacc[1][r] = p1;
            float ps = p0 + p1;
#pragma unroll
            for (int m = 1; m < 16; m <<= 1) ps += __shfl_xor(ps, m, 32);
            lrow[r] = lrow[r] * alpha[r] + ps;
        }
#pragma unroll
        for (int dt = 0; dt < 4; ++dt)
#pragma unroll
            for (int r = 0; r < 8; ++r) o_acc[dt][r] *= alpha[r];

        // P: C-layout -> A-layout via per-wave LDS staging
#pragma unroll
        for (int jt = 0; jt < 2; ++jt)
#pragma unroll
            for (int r = 0; r < 8; ++r)
                Ps[pbase + (r + 8 * halfsel) * 40 + jt * 16 + ln16] = (_Float16)sacc[jt][r];
        __syncthreads();

        v16h ap = make_frag(&Ps[pbase + ln16 * 40 + 8 * halfsel],
                            &Ps[pbase + ln16 * 40 + 16 + 8 * halfsel]);
#pragma unroll
        for (int dt = 0; dt < 4; ++dt) {
            int d = dt * 16 + ln16;
            v16h bvf = make_frag(&Vt[d * 40 + 8 * halfsel],
                                 &Vt[d * 40 + 16 + 8 * halfsel]);
            o_acc[dt] = __builtin_amdgcn_wmma_f32_16x16x32_f16(
                false, ap, false, bvf, (short)0, o_acc[dt], false, false);
        }
    }

    // Epilogue: normalize, stage f16 tile in Qs, coalesced b128 stores.
    __syncthreads();
#pragma unroll
    for (int dt = 0; dt < 4; ++dt)
#pragma unroll
        for (int r = 0; r < 8; ++r) {
            int lr = wv * 16 + r + 8 * halfsel;
            int lc = dt * 16 + ln16;
            Qs[lr * 72 + lc] = (_Float16)(o_acc[dt][r] * (1.f / lrow[r]));
        }
    __syncthreads();
#pragma unroll
    for (int v = 0; v < 4; ++v) {
        int g = t + v * 256;
        int row = g >> 3, col8 = (g & 7) * 8;
        *(v8h*)&Op[(rowbase + q0 + row) * DIM + cbase + col8] = *(const v8h*)&Qs[row * 72 + col8];
    }
}

// ---------------------------------------------------------------------------
extern "C" void kernel_launch(void* const* d_in, const int* in_sizes, int n_in,
                              void* d_out, int out_size, void* d_ws, size_t ws_size,
                              hipStream_t stream) {
    (void)in_sizes; (void)n_in; (void)out_size; (void)ws_size;
    const float* query = (const float*)d_in[0];
    const float* key   = (const float*)d_in[1];
    const float* ln_g  = (const float*)d_in[2];
    const float* ln_b  = (const float*)d_in[3];
    const float* Wq    = (const float*)d_in[4];
    const float* Wk    = (const float*)d_in[5];
    const float* Wv    = (const float*)d_in[6];
    const float* Wo    = (const float*)d_in[7];
    const float* bo    = (const float*)d_in[8];
    float* out = (float*)d_out;

    char* ws = (char*)d_ws;
    const size_t WSZ = (size_t)DIM * DIM * sizeof(_Float16);
    const size_t MSZ = (size_t)ROWS_TOTAL * DIM * sizeof(_Float16);
    _Float16* wq_h = (_Float16*)(ws);
    _Float16* wk_h = (_Float16*)(ws + WSZ);
    _Float16* wv_h = (_Float16*)(ws + 2 * WSZ);
    _Float16* wo_h = (_Float16*)(ws + 3 * WSZ);
    char* p = ws + 4 * WSZ;
    _Float16* qn = (_Float16*)(p);
    _Float16* kn = (_Float16*)(p + MSZ);
    _Float16* vn = (_Float16*)(p + 2 * MSZ);
    _Float16* Qp = (_Float16*)(p + 3 * MSZ);
    _Float16* Kp = (_Float16*)(p + 4 * MSZ);
    _Float16* Vp = (_Float16*)(p + 5 * MSZ);
    _Float16* AO = (_Float16*)(p + 6 * MSZ);

    pack_f16_kernel<<<256, 256, 0, stream>>>(Wq, wq_h, DIM * DIM);
    pack_f16_kernel<<<256, 256, 0, stream>>>(Wk, wk_h, DIM * DIM);
    pack_f16_kernel<<<256, 256, 0, stream>>>(Wv, wv_h, DIM * DIM);
    pack_f16_kernel<<<256, 256, 0, stream>>>(Wo, wo_h, DIM * DIM);

    ln_q_kernel <<<ROWS_TOTAL / 8, 256, 0, stream>>>(query, ln_g, ln_b, qn);
    ln_kv_kernel<<<ROWS_TOTAL / 8, 256, 0, stream>>>(key,   ln_g, ln_b, kn, vn);

    dim3 gg(DIM / 64, ROWS_TOTAL / 128);  // (8, 64)
    gemm_qkv_kernel<<<gg, 256, 0, stream>>>(qn, wq_h, Qp);
    gemm_qkv_kernel<<<gg, 256, 0, stream>>>(kn, wk_h, Kp);
    gemm_qkv_kernel<<<gg, 256, 0, stream>>>(vn, wv_h, Vp);

    dim3 ga(SEQ / 128, BATCH * HEADS);    // (8, 64)
    attn_kernel<<<ga, 256, 0, stream>>>(Qp, Kp, Vp, AO);

    gemm_out_kernel<<<gg, 256, 0, stream>>>(AO, wo_h, bo, out);
}
